// FlashAttentionBlock_29463475651359
// MI455X (gfx1250) — matrix operand order repaired and checked by reading, not compiled, not run on hardware
//
#include <hip/hip_runtime.h>

// ---------------------------------------------------------------------------
// Tiled-attention block for MI455X (gfx1250, wave32, WMMA + TDM/async-LDS).
// Pipeline (all on `stream`):
//   1) q  = x @ Wq^T + bq   (bf16 out, row-major)        [WMMA GEMM]
//   2) k  = x @ Wk^T + bk   (bf16 out, row-major)        [WMMA GEMM]
//   3) vT = (x @ Wv^T + bv)^T (bf16 out, [D][B*S])       [WMMA GEMM, transposed store]
//   4) attn: per-kv-tile softmax attention (bf16 out)    [WMMA; Q staged via TDM]
//   5) out = attn @ Wo^T + bo (fp32 out -> d_out)        [WMMA GEMM; A staged via
//                                                         global_load_async_to_lds]
// Workspace requirement: 4 * 8192*1024 bf16 = 64 MB in d_ws.
// ---------------------------------------------------------------------------

typedef __attribute__((ext_vector_type(16))) __bf16 v16bf;
typedef __attribute__((ext_vector_type(8)))  __bf16 v8bf;
typedef __attribute__((ext_vector_type(8)))  float  v8f;
typedef __attribute__((ext_vector_type(16))) float  f32x16;
typedef unsigned int u32x4 __attribute__((ext_vector_type(4)));
typedef unsigned int u32x8 __attribute__((ext_vector_type(8)));

#define HIDDEN   1024
#define SEQ      4096
#define BATCH    2
#define MTOT     (BATCH * SEQ)      // 8192 rows
#define KV_TILE  128
#define NKV      (SEQ / KV_TILE)    // 32 kv tiles

__device__ __forceinline__ v8f zero8() {
    v8f z;
#pragma unroll
    for (int i = 0; i < 8; ++i) z[i] = 0.0f;
    return z;
}

__device__ __forceinline__ v16bf cat8(v8bf lo, v8bf hi) {
    return __builtin_shufflevector(lo, hi, 0, 1, 2, 3, 4, 5, 6, 7,
                                           8, 9, 10, 11, 12, 13, 14, 15);
}

// A-fragment (16x32 bf16, contiguous K layout matching B):
// lane group g = lane/16 holds K [k0+g*16, +16).
__device__ __forceinline__ v16bf load_a_frag(const __bf16* p, int stride,
                                             int row, int k0, int g) {
    const __bf16* q = p + row * stride + k0 + g * 16;
    v8bf lo = *(const v8bf*)(q);
    v8bf hi = *(const v8bf*)(q + 8);
    return cat8(lo, hi);
}

__device__ __forceinline__ v8f wmma_bf16(v16bf a, v16bf b, v8f c) {
    return __builtin_amdgcn_wmma_f32_16x16x32_bf16(
        /*neg_a=*/false, a, /*neg_b=*/false, b,
        /*c_mod=*/(short)0, c, /*reuse_a=*/false, /*reuse_b=*/false);
}

// Low 32 bits of a generic pointer to LDS = workgroup-relative LDS byte offset
// (aperture bits live in [63:32], ISA 10.2 aperture table).
__device__ __forceinline__ unsigned lds_offset(const void* p) {
    return (unsigned)(unsigned long long)p;
}

// ---------------------------------------------------------------------------
// GEMM: C[m][n] = sum_k A[m][k] * W[n][k] + bias[n]
//   A_IS_F32:  A read as fp32 (packed-converted to bf16 in LDS) or bf16
//              (staged register-free via GLOBAL_LOAD_ASYNC_TO_LDS, ASYNCcnt)
//   STORE_MODE: 0 = bf16 row-major [M][N], 1 = bf16 transposed [N][M],
//               2 = fp32 row-major [M][N]
// Block = 128x128 C tile, 8 waves (4 along M x 2 along N), K-step 32.
// ---------------------------------------------------------------------------
template <bool A_IS_F32, int STORE_MODE>
__global__ __launch_bounds__(256) void gemm_wmma_kernel(
    const void* __restrict__ Av, const float* __restrict__ W,
    const float* __restrict__ bias, void* __restrict__ Cv,
    int M, int K, int N)
{
    constexpr int KS   = 32;
    constexpr int LSTR = 48;  // padded LDS stride (elements): 96B, 32B-aligned
    __shared__ __bf16 lA[128 * LSTR];
    __shared__ __bf16 lB[128 * LSTR];

    const int tid  = threadIdx.x;
    const int lane = tid & 31;
    const int wid  = tid >> 5;
    const int g    = lane >> 4;     // lane group (0/1)
    const int ln   = lane & 15;
    const int wm   = wid & 3;       // wave M position (0..3) * 32 rows
    const int wn   = wid >> 2;      // wave N position (0..1) * 64 cols
    const int mb   = blockIdx.y * 128;
    const int nb   = blockIdx.x * 128;

    const float*  Af = (const float*)Av;
    const __bf16* Ab = (const __bf16*)Av;

    v8f acc[2][4];
#pragma unroll
    for (int mt = 0; mt < 2; ++mt)
#pragma unroll
        for (int nt = 0; nt < 4; ++nt) acc[mt][nt] = zero8();

    const int srow  = tid >> 1;  // 0..127 staging row
    const int shalf = tid & 1;   // which 16-element K chunk

    for (int kb = 0; kb < K; kb += KS) {
        __syncthreads();
        // ---- stage A tile (128 x 32) as bf16 ----
        if constexpr (A_IS_F32) {
            const f32x16 src =
                *(const f32x16*)(Af + (size_t)(mb + srow) * K + kb + shalf * 16);
            *(v16bf*)&lA[srow * LSTR + shalf * 16] =
                __builtin_convertvector(src, v16bf);  // packed v_cvt_pk_bf16_f32
        } else {
            // Register-free copy: 2x 16B chunks per thread, tracked by ASYNCcnt.
            const __bf16* src = Ab + (size_t)(mb + srow) * K + kb + shalf * 16;
            const unsigned l0 = lds_offset(&lA[srow * LSTR + shalf * 16]);
            asm volatile("global_load_async_to_lds_b128 %0, %1, off"
                         :: "v"(l0), "v"(src) : "memory");
            asm volatile("global_load_async_to_lds_b128 %0, %1, off"
                         :: "v"(l0 + 16u), "v"(src + 8) : "memory");
        }
        // ---- stage W tile (128 x 32) as bf16 ----
        {
            const f32x16 src =
                *(const f32x16*)(W + (size_t)(nb + srow) * K + kb + shalf * 16);
            *(v16bf*)&lB[srow * LSTR + shalf * 16] =
                __builtin_convertvector(src, v16bf);
        }
        if constexpr (!A_IS_F32)
            asm volatile("s_wait_asynccnt 0x0" ::: "memory");
        __syncthreads();

        // prefetch next K tiles into cache (emits global_prefetch_b8)
        if (kb + KS < K) {
            if constexpr (A_IS_F32)
                __builtin_prefetch(Af + (size_t)(mb + srow) * K + kb + KS + shalf * 16, 0, 1);
            else
                __builtin_prefetch(Ab + (size_t)(mb + srow) * K + kb + KS + shalf * 16, 0, 1);
            __builtin_prefetch(W + (size_t)(nb + srow) * K + kb + KS + shalf * 16, 0, 1);
        }

        // ---- compute: 2 A-frags x 4 B-frags = 8 WMMAs per K-step ----
        v16bf af[2], bf[4];
#pragma unroll
        for (int mt = 0; mt < 2; ++mt)
            af[mt] = load_a_frag(lA, LSTR, wm * 32 + mt * 16 + ln, 0, g);
#pragma unroll
        for (int nt = 0; nt < 4; ++nt)  // B[k][n]=W[n][k]: contiguous 16 K at fixed n
            bf[nt] = *(const v16bf*)&lB[(wn * 64 + nt * 16 + ln) * LSTR + g * 16];
#pragma unroll
        for (int mt = 0; mt < 2; ++mt)
#pragma unroll
            for (int nt = 0; nt < 4; ++nt)
                acc[mt][nt] = wmma_bf16(af[mt], bf[nt], acc[mt][nt]);
    }

    // ---- epilogue: bias + store ----
#pragma unroll
    for (int mt = 0; mt < 2; ++mt) {
#pragma unroll
        for (int nt = 0; nt < 4; ++nt) {
            const int n  = nb + wn * 64 + nt * 16 + ln;
            const int m0 = mb + wm * 32 + mt * 16 + 8 * g;  // C-frag: M = r + 8*(lane/16)
            const float bv = bias[n];
            if constexpr (STORE_MODE == 1) {
                v8bf o;
#pragma unroll
                for (int r = 0; r < 8; ++r) o[r] = (__bf16)(acc[mt][nt][r] + bv);
                *(v8bf*)((__bf16*)Cv + (size_t)n * M + m0) = o;  // C^T, contiguous in m
            } else if constexpr (STORE_MODE == 0) {
                __bf16* C = (__bf16*)Cv;
#pragma unroll
                for (int r = 0; r < 8; ++r)
                    C[(size_t)(m0 + r) * N + n] = (__bf16)(acc[mt][nt][r] + bv);
            } else {
                float* C = (float*)Cv;
#pragma unroll
                for (int r = 0; r < 8; ++r)
                    C[(size_t)(m0 + r) * N + n] = acc[mt][nt][r] + bv;
            }
        }
    }
}

// ---------------------------------------------------------------------------
// Attention: per kv-tile softmax.
// Block = (batch b, 16 query rows), 8 waves.
//   Q tile (16x1024 bf16 = 32KB) staged by the Tensor Data Mover: one
//   tensor_load_to_lds with a 2-group D# (1-D tile of 16384 bf16 elements,
//   pad_enable: 4 DWORDs inserted every 256 DWORDs -> 8 bf16 pad per 512
//   elements == padded LDS stride 1040 with a mid-row pad at dk==512).
//   Phase A: wave w computes scores for keys [w*16, w*16+16) over full D=1024.
//   Softmax: 16 row-threads normalize each 128-key row; accumulate norm.
//   Phase B: wave w accumulates out columns [w*128, w*128+128) from w @ V_t,
//            reading V^T so per-lane B-fragments are contiguous.
// ---------------------------------------------------------------------------
__global__ __launch_bounds__(256) void attn_tile_kernel(
    const __bf16* __restrict__ Q, const __bf16* __restrict__ Kmat,
    const __bf16* __restrict__ VT, __bf16* __restrict__ O)
{
    constexpr int QB   = 16;
    constexpr int QSTR = 1040;  // 1024 + 2x8 mid/end pads from TDM pad feature
    constexpr int SSTR = 132;   // fp32 scores stride
    constexpr int WSTR = 144;   // bf16 weights stride (288B: 32B-aligned)
    __shared__ __bf16 lQ[QB * QSTR];
    __shared__ float  lS[QB * SSTR];
    __shared__ __bf16 lW[QB * WSTR];
    __shared__ float  lNorm[QB];

    const int tid  = threadIdx.x;
    const int lane = tid & 31;
    const int wid  = tid >> 5;   // 0..7
    const int g    = lane >> 4;
    const int ln   = lane & 15;

    const int b  = blockIdx.x >> 8;    // 256 query blocks per batch
    const int qb = blockIdx.x & 255;
    const size_t qrow0 = (size_t)b * SEQ + (size_t)qb * QB;

    // ---- stage Q block via TDM (wave 0 issues; EXEC-independent DMA) ----
    if (wid == 0) {
        const unsigned long long ga =
            (unsigned long long)(const void*)(Q + qrow0 * HIDDEN);
        const unsigned ldsa = lds_offset(&lQ[0]);
        u32x4 g0;
        g0[0] = 1u;                                   // count=1, load, user D#
        g0[1] = ldsa;                                 // lds_addr (bytes)
        g0[2] = (unsigned)(ga & 0xFFFFFFFFull);       // global_addr[31:0]
        g0[3] = (unsigned)((ga >> 32) & 0x1FFFFFFull) // global_addr[56:32]
                | 0x80000000u;                        // type=2 ("image")
        u32x8 g1;
        g1[0] = (1u << 16)     // data_size = 2 bytes
              | (1u << 20)     // pad_enable
              | (7u << 22)     // pad_interval: every 256 DWORDs
              | (3u << 25);    // pad_amount: 4 DWORDs (= 8 bf16)
        g1[1] = (16384u & 0xFFFFu) << 16;             // tensor_dim0 lo16
        g1[2] = (16384u >> 16) | (1u << 16);          // tensor_dim0 hi16 | tensor_dim1
        g1[3] = (16384u << 16);                       // tile_dim0 = 16384 (1-D tile)
        g1[4] = 0u;                                   // tile_dim1=0, tile_dim2=0
        g1[5] = 16384u;                               // tensor_dim0_stride lo32
        g1[6] = 0u;
        g1[7] = 0u;
        asm volatile("tensor_load_to_lds %0, %1" :: "s"(g0), "s"(g1) : "memory");
        __builtin_amdgcn_s_wait_tensorcnt(0);
    }
    if (tid < QB) lNorm[tid] = 0.0f;
    __syncthreads();

    v8f acc[8];
#pragma unroll
    for (int dt = 0; dt < 8; ++dt) acc[dt] = zero8();

    const float scale = 0.03125f;  // 1/sqrt(1024)

    for (int t = 0; t < NKV; ++t) {
        // ---- Phase A: scores[16q x 16k] for this wave's key strip ----
        v8f sc = zero8();
        const __bf16* kbase =
            Kmat + ((size_t)b * SEQ + (size_t)t * KV_TILE + wid * 16 + ln) * HIDDEN;
        __builtin_prefetch(kbase + HIDDEN * 16, 0, 1);
        for (int dk = 0; dk < HIDDEN; dk += 32) {
            const int dkp = dk + ((dk >> 9) << 3);            // TDM mid-row pad skip
            v16bf a  = load_a_frag(lQ, QSTR, ln, dkp, g);     // m = lane%16
            v16bf bk = *(const v16bf*)(kbase + dk + g * 16);  // n = key, K contig
            sc = wmma_bf16(a, bk, sc);
        }
#pragma unroll
        for (int r = 0; r < 8; ++r)
            lS[(r + 8 * g) * SSTR + wid * 16 + ln] = sc[r] * scale;
        __syncthreads();

        // ---- per-tile softmax over the 128 keys, one thread per query row ----
        if (tid < QB) {
            float* row = &lS[tid * SSTR];
            float mx = row[0];
            for (int i = 1; i < KV_TILE; ++i) mx = fmaxf(mx, row[i]);
            float s = 0.0f;
            for (int i = 0; i < KV_TILE; ++i) {
                float e = __expf(row[i] - mx);
                row[i] = e;
                s += e;
            }
            const float inv = 1.0f / s;
            for (int i = 0; i < KV_TILE; ++i)
                lW[tid * WSTR + i] = (__bf16)(row[i] * inv);
            lNorm[tid] += 1.0f;  // sum of normalized per-tile weights
        }
        __syncthreads();

        // ---- Phase B: acc += w[16 x 128] @ V_t[128 x 128d] via V^T ----
        const __bf16* vtb = VT + (size_t)b * SEQ + (size_t)t * KV_TILE;
        for (int kk = 0; kk < KV_TILE; kk += 32) {
            v16bf a = load_a_frag(lW, WSTR, ln, kk, g);
#pragma unroll
            for (int dt = 0; dt < 8; ++dt) {
                const int n = wid * 128 + dt * 16 + ln;
                v16bf bv = *(const v16bf*)(vtb + (size_t)n * MTOT + kk + g * 16);
                acc[dt] = wmma_bf16(a, bv, acc[dt]);
            }
        }
    }
    __syncthreads();

    // ---- epilogue: divide by norm, store bf16 attention output ----
#pragma unroll
    for (int dt = 0; dt < 8; ++dt) {
#pragma unroll
        for (int r = 0; r < 8; ++r) {
            const int q = r + 8 * g;
            const float nv = lNorm[q] + 1e-6f;
            const float o = acc[dt][r] / nv;
            O[(qrow0 + q) * HIDDEN + wid * 128 + dt * 16 + ln] = (__bf16)o;
        }
    }
}

// ---------------------------------------------------------------------------
// Host launcher
// ---------------------------------------------------------------------------
extern "C" void kernel_launch(void* const* d_in, const int* in_sizes, int n_in,
                              void* d_out, int out_size, void* d_ws, size_t ws_size,
                              hipStream_t stream) {
    const float* batch = (const float*)d_in[0];
    const float* Wq    = (const float*)d_in[1];
    const float* bq    = (const float*)d_in[2];
    const float* Wk    = (const float*)d_in[3];
    const float* bk    = (const float*)d_in[4];
    const float* Wv    = (const float*)d_in[5];
    const float* bv    = (const float*)d_in[6];
    const float* Wo    = (const float*)d_in[7];
    const float* bo    = (const float*)d_in[8];
    float* out = (float*)d_out;

    // Workspace layout (bf16): q | k | vT | attn  (4 x 16 MB = 64 MB)
    const size_t elems = (size_t)MTOT * HIDDEN;
    __bf16* q_ws    = (__bf16*)d_ws;
    __bf16* k_ws    = q_ws + elems;
    __bf16* vT_ws   = k_ws + elems;
    __bf16* attn_ws = vT_ws + elems;

    dim3 gblk(256);
    dim3 ggrid(HIDDEN / 128, MTOT / 128);  // (8, 64)

    // Projections: fp32 input, bf16 output (V stored transposed as [D][B*S]).
    gemm_wmma_kernel<true, 0><<<ggrid, gblk, 0, stream>>>(
        batch, Wq, bq, q_ws, MTOT, HIDDEN, HIDDEN);
    gemm_wmma_kernel<true, 0><<<ggrid, gblk, 0, stream>>>(
        batch, Wk, bk, k_ws, MTOT, HIDDEN, HIDDEN);
    gemm_wmma_kernel<true, 1><<<ggrid, gblk, 0, stream>>>(
        batch, Wv, bv, vT_ws, MTOT, HIDDEN, HIDDEN);

    // Attention (per-kv-tile softmax), bf16 output.
    attn_tile_kernel<<<dim3(MTOT / 16), gblk, 0, stream>>>(
        q_ws, k_ws, vT_ws, attn_ws);

    // Output projection: bf16 input, fp32 output.
    gemm_wmma_kernel<false, 2><<<ggrid, gblk, 0, stream>>>(
        attn_ws, Wo, bo, out, MTOT, HIDDEN, HIDDEN);
}
